// MLPTexture3D_25185688224084
// MI455X (gfx1250) — matrix operand order repaired and checked by reading, not compiled, not run on hardware
//
#include <hip/hip_runtime.h>

typedef __attribute__((ext_vector_type(16))) _Float16 v16h;
typedef __attribute__((ext_vector_type(8)))  _Float16 v8h;
typedef __attribute__((ext_vector_type(2)))  _Float16 h2;
typedef __attribute__((ext_vector_type(8)))  float    v8f;

#define WMMA16(a, b, c) \
  __builtin_amdgcn_wmma_f32_16x16x32_f16(false, (a), false, (b), (short)0, (c), false, false)

__device__ __forceinline__ v16h make_frag(v8h lo, v8h hi) {
  v16h r;
#pragma unroll
  for (int i = 0; i < 8; ++i) { r[i] = lo[i]; r[8 + i] = hi[i]; }
  return r;
}

// A fragment: 16x32 f16 tile stored row-major in LDS with stride 32 halfs.
// ISA layout: lane L -> row M = L&15; halfs 0..7 = K 8h..8h+7, halfs 8..15 = K 16+8h..16+8h+7 (h = L>>4).
__device__ __forceinline__ v16h load_afrag(const _Float16* base, int lane) {
  const int m = lane & 15;
  const int h = lane >> 4;
  const _Float16* rp = base + m * 32 + 8 * h;
  v8h lo = *(const v8h*)(rp);       // K = 8h .. 8h+7
  v8h hi = *(const v8h*)(rp + 16);  // K = 16+8h .. 16+8h+7
  return make_frag(lo, hi);
}

// B fragment: 32x16 f16 tile; weights staged in LDS as [N][K] row-major (stride 32 halfs).
// ISA layout: lane L -> N = L&15; halfs 0..7 = K 8h..8h+7, halfs 8..15 = K 16+8h..16+8h+7 (h = L>>4).
__device__ __forceinline__ v16h load_bfrag(const _Float16* sB, int ncol, int lane) {
  const _Float16* rp = sB + ncol * 32 + (lane >> 4) * 8;
  v8h lo = *(const v8h*)(rp);
  v8h hi = *(const v8h*)(rp + 16);
  return make_frag(lo, hi);
}

// f32->f16 convert (v_cvt_pk_f16_f32), then packed f16 ReLU (v_pk_max_num_f16,
// no canonicalize needed since cvt results are canonical), then the two halves
// go to LDS as ds_store_b16 / ds_store_b16_d16_hi from the same VGPR.
__device__ __forceinline__ void store_relu_tile(_Float16* hrow, v8f ca, v8f cb,
                                                int nIdx, int hsel) {
#pragma unroll
  for (int r = 0; r < 8; ++r) {
    const int M = r + 8 * hsel;
    h2 v;
    v[0] = (_Float16)ca[r];
    v[1] = (_Float16)cb[r];
    const h2 z = {(_Float16)0.0f, (_Float16)0.0f};
    v = __builtin_elementwise_max(v, z);
    hrow[M * 32 + nIdx] = v[0];
    hrow[M * 32 + 16 + nIdx] = v[1];
  }
}

__global__ __launch_bounds__(256) void mlp_tex3d_kernel(
    const float* __restrict__ texc, const float* __restrict__ table,
    const float* __restrict__ W0, const float* __restrict__ W1,
    const float* __restrict__ W2, const float* __restrict__ minv,
    const float* __restrict__ maxv, float* __restrict__ out, int nPts) {
  __shared__ __align__(16) _Float16 sB0[32 * 32];    // layer0 weights [N=32][K=32] (K 10..31 zero)
  __shared__ __align__(16) _Float16 sB1[32 * 32];    // layer1 weights [N=32][K=32]
  __shared__ __align__(16) _Float16 sB2[16 * 32];    // layer2 weights [N=16][K=32] (N 3..15 zero)
  __shared__ __align__(16) _Float16 sFeat[256 * 32]; // per-point features, padded to K=32
  __shared__ __align__(16) _Float16 sH[8 * 2 * 16 * 32]; // per-wave hidden tiles, one per 16-row subtile
  __shared__ __align__(16) float sOut[8 * 96];       // per-wave output staging 32x3

  const int tid = threadIdx.x;
  const int lane = tid & 31;
  const int wave = tid >> 5;
  const int nIdx = lane & 15;
  const int hsel = lane >> 4;

  // ---- stage weights into LDS as f16 [N][K] ----
  for (int i = tid; i < 32 * 32; i += 256) {
    const int n = i >> 5, k = i & 31;
    sB0[i] = (k < 10) ? (_Float16)W0[n * 10 + k] : (_Float16)0.0f;
  }
  for (int i = tid; i < 32 * 32; i += 256) sB1[i] = (_Float16)W1[i];  // W1 row-major == [N][K]
  for (int i = tid; i < 16 * 32; i += 256) {
    const int n = i >> 5, k = i & 31;
    sB2[i] = (n < 3) ? (_Float16)W2[n * 32 + k] : (_Float16)0.0f;
  }
  // zero the K=10..31 pad of this thread's feature row (written once, never touched again)
#pragma unroll
  for (int c = 10; c < 32; ++c) sFeat[tid * 32 + c] = (_Float16)0.0f;
  __syncthreads();

  const v16h b0a = load_bfrag(sB0, nIdx, lane);
  const v16h b0b = load_bfrag(sB0, 16 + nIdx, lane);
  const v16h b1a = load_bfrag(sB1, nIdx, lane);
  const v16h b1b = load_bfrag(sB1, 16 + nIdx, lane);
  const v16h b2f = load_bfrag(sB2, nIdx, lane);

  // loop-invariant output affine params (only lanes with nIdx<3 use them)
  const float mn = (nIdx < 3) ? minv[nIdx] : 0.0f;
  const float mx = (nIdx < 3) ? maxv[nIdx] : 0.0f;
  const float scl = mx - mn;

  const int RESL[5] = {16, 23, 33, 48, 70};
  const int OFFSL[5] = {0, 4913, 18737, 58041, 175690};
  const float2* tb = (const float2*)table;
  _Float16* frow = &sFeat[tid * 32];
  _Float16* hrow0 = &sH[wave * 1024];        // subtile 0 hidden buffer [16][32]
  _Float16* hrow1 = hrow0 + 512;             // subtile 1 hidden buffer [16][32]
  float* so = &sOut[wave * 96];

  const long stride = (long)gridDim.x * 256;
  for (long base = (long)blockIdx.x * 256; base < nPts; base += stride) {
    const long p = base + tid;
    const long pc = (p < nPts) ? p : (long)(nPts - 1);
    const float cx = texc[pc * 3 + 0];
    const float cy = texc[pc * 3 + 1];
    const float cz = texc[pc * 3 + 2];
    // (x - bbox_min) / (bbox_max - bbox_min), clipped to [0,1]
    const float ux = fminf(fmaxf((cx - 0.6f) * (-1.0f / 1.4f), 0.0f), 1.0f);
    const float uy = fminf(fmaxf((cy - 0.6f) * (-1.0f / 1.8f), 0.0f), 1.0f);
    const float uz = fminf(fmaxf((cz - 0.2f) * (-1.0f / 0.4f), 0.0f), 1.0f);

    // ---- multilevel trilinear encoding (table is L2-resident: 4.27 MB) ----
#pragma unroll
    for (int l = 0; l < 5; ++l) {
      const float rf = (float)RESL[l];
      const int g = RESL[l] + 1;
      const float px = ux * rf, py = uy * rf, pz = uz * rf;
      const float fx0 = fminf(floorf(px), rf - 1.0f);
      const float fy0 = fminf(floorf(py), rf - 1.0f);
      const float fz0 = fminf(floorf(pz), rf - 1.0f);
      const float fx = px - fx0, fy = py - fy0, fz = pz - fz0;
      const int ix = (int)fx0, iy = (int)fy0, iz = (int)fz0;
      const float wx[2] = {1.0f - fx, fx};
      const float wy[2] = {1.0f - fy, fy};
      const float wz[2] = {1.0f - fz, fz};
      const int b000 = (ix * g + iy) * g + iz + OFFSL[l];
      float ax = 0.0f, ay = 0.0f;
#pragma unroll
      for (int dx = 0; dx < 2; ++dx)
#pragma unroll
        for (int dy = 0; dy < 2; ++dy)
#pragma unroll
          for (int dz = 0; dz < 2; ++dz) {
            const int idx = b000 + (dx * g + dy) * g + dz;
            const float w = wx[dx] * wy[dy] * wz[dz];
            const float2 t = tb[idx];
            ax = fmaf(w, t.x, ax);
            ay = fmaf(w, t.y, ay);
          }
      frow[2 * l + 0] = (_Float16)ax;
      frow[2 * l + 1] = (_Float16)ay;
    }
    asm volatile("s_wait_dscnt 0" ::: "memory");  // feature rows visible wave-wide

    // ---- MLP: both 16-row subtiles processed layer-by-layer so the second
    //      tile's WMMAs fill the first tile's WMMA->VALU hazard window ----
    // Layer 0
    const v16h a0_0 = load_afrag(&sFeat[(wave * 32 + 0) * 32], lane);
    const v16h a0_1 = load_afrag(&sFeat[(wave * 32 + 16) * 32], lane);
    v8f c0a0 = {}, c0b0 = {}, c0a1 = {}, c0b1 = {};
    c0a0 = WMMA16(a0_0, b0a, c0a0);
    c0b0 = WMMA16(a0_0, b0b, c0b0);
    c0a1 = WMMA16(a0_1, b0a, c0a1);
    c0b1 = WMMA16(a0_1, b0b, c0b1);
    store_relu_tile(hrow0, c0a0, c0b0, nIdx, hsel);
    store_relu_tile(hrow1, c0a1, c0b1, nIdx, hsel);
    asm volatile("s_wait_dscnt 0" ::: "memory");

    // Layer 1 (K=32 == native f16 WMMA K)
    const v16h a1_0 = load_afrag(hrow0, lane);
    const v16h a1_1 = load_afrag(hrow1, lane);
    v8f c1a0 = {}, c1b0 = {}, c1a1 = {}, c1b1 = {};
    c1a0 = WMMA16(a1_0, b1a, c1a0);
    c1b0 = WMMA16(a1_0, b1b, c1b0);
    c1a1 = WMMA16(a1_1, b1a, c1a1);
    c1b1 = WMMA16(a1_1, b1b, c1b1);
    store_relu_tile(hrow0, c1a0, c1b0, nIdx, hsel);
    store_relu_tile(hrow1, c1a1, c1b1, nIdx, hsel);
    asm volatile("s_wait_dscnt 0" ::: "memory");

    // Layer 2 (N padded 3->16)
    const v16h a2_0 = load_afrag(hrow0, lane);
    const v16h a2_1 = load_afrag(hrow1, lane);
    v8f c2_0 = {}, c2_1 = {};
    c2_0 = WMMA16(a2_0, b2f, c2_0);
    c2_1 = WMMA16(a2_1, b2f, c2_1);

    // sigmoid * (max-min) + min, staged to LDS for coalesced writeout
    if (nIdx < 3) {
#pragma unroll
      for (int r = 0; r < 8; ++r) {
        const float s0 = 1.0f / (1.0f + __expf(-c2_0[r]));
        const float s1 = 1.0f / (1.0f + __expf(-c2_1[r]));
        so[(r + 8 * hsel) * 3 + nIdx] = fmaf(s0, scl, mn);
        so[48 + (r + 8 * hsel) * 3 + nIdx] = fmaf(s1, scl, mn);
      }
    }
    asm volatile("s_wait_dscnt 0" ::: "memory");

    // coalesced 96-float (32 points x 3 ch) tile writeout: 3 b32 stores, all lanes active
    const long row0 = base + wave * 32;
    const long lim = (long)nPts * 3;
#pragma unroll
    for (int j = 0; j < 3; ++j) {
      const long e = row0 * 3 + 32 * j + lane;
      if (e < lim) out[e] = so[32 * j + lane];
    }
  }
}

extern "C" void kernel_launch(void* const* d_in, const int* in_sizes, int n_in,
                              void* d_out, int out_size, void* d_ws, size_t ws_size,
                              hipStream_t stream) {
  const float* texc = (const float*)d_in[0];
  const float* table = (const float*)d_in[1];
  const float* W0 = (const float*)d_in[2];
  const float* W1 = (const float*)d_in[3];
  const float* W2 = (const float*)d_in[4];
  const float* minv = (const float*)d_in[5];
  const float* maxv = (const float*)d_in[6];
  // d_in[7] = frame_id (unused)
  const int nPts = in_sizes[0] / 3;
  float* out = (float*)d_out;

  int nChunks = (nPts + 255) / 256;
  int grid = nChunks < 8192 ? nChunks : 8192;
  if (grid < 1) grid = 1;
  mlp_tex3d_kernel<<<grid, 256, 0, stream>>>(texc, table, W0, W1, W2, minv, maxv, out, nPts);
}
